// SceneInteractionNet_51608327028955
// MI455X (gfx1250) — matrix operand order repaired and checked by reading, not compiled, not run on hardware
//
#include <hip/hip_runtime.h>

typedef _Float16 v16h __attribute__((ext_vector_type(16)));
typedef _Float16 h8   __attribute__((ext_vector_type(8)));
typedef float    v8f  __attribute__((ext_vector_type(8)));
typedef int      v4i  __attribute__((ext_vector_type(4)));

#if defined(__has_builtin)
#  if __has_builtin(__builtin_amdgcn_global_load_async_to_lds_b128)
#    define USE_ASYNC_LDS 1
#  endif
#  if __has_builtin(__builtin_amdgcn_s_wait_asynccnt)
#    define HAVE_WAIT_ASYNC_BUILTIN 1
#  endif
#endif

// ---------- helpers ----------
__device__ __forceinline__ unsigned fmono(float x) {
  unsigned u = __float_as_uint(x);
  return (u & 0x80000000u) ? ~u : (u | 0x80000000u);
}
__device__ __forceinline__ float funmono(unsigned u) {
  unsigned b = (u & 0x80000000u) ? (u & 0x7fffffffu) : ~u;
  return __uint_as_float(b);
}
__device__ __forceinline__ float nan0(float x) { return (x != x) ? 0.0f : x; }

__device__ __forceinline__ void zero_h8(_Float16* p) {
  #pragma unroll
  for (int i = 0; i < 8; ++i) p[i] = (_Float16)0.0f;
}

// A fragment (16x32 f16): lanes 0-15 row M=lane, halves0-15:K=0..15
//                         lanes16-31 row M=lane-16, halves0-15:K=16..31
__device__ __forceinline__ v16h load_a_frag(const _Float16* act, int lane, int kc, int AS) {
  const int row = lane & 15, grp = lane >> 4;
  const _Float16* p = act + row * AS + kc * 32 + grp * 16;
  union { v16h v; h8 h[2]; } u;
  u.h[0] = *(const h8*)(p);
  u.h[1] = *(const h8*)(p + 8);
  return u.v;
}
// B fragments are pre-swizzled: contiguous 16 halves per (frag,lane)
__device__ __forceinline__ v16h load_b_frag(const _Float16* wlds, int frag, int lane) {
  return *(const v16h*)(wlds + ((frag * 32 + lane) << 4));
}

template<int KC, int NT, bool RELU>
__device__ __forceinline__ void run_layer(const _Float16* act, int AS,
                                          const _Float16* wlds,
                                          const float* __restrict__ bias,
                                          int lane, v8f* acc) {
  const int col = lane & 15;
  #pragma unroll
  for (int nt = 0; nt < NT; ++nt) {
    float b = bias[nt * 16 + col];
    #pragma unroll
    for (int i = 0; i < 8; ++i) acc[nt][i] = b;
  }
  #pragma unroll
  for (int kc = 0; kc < KC; ++kc) {
    v16h a = load_a_frag(act, lane, kc, AS);
    #pragma unroll
    for (int nt = 0; nt < NT; ++nt) {
      v16h bm = load_b_frag(wlds, kc * NT + nt, lane);
      acc[nt] = __builtin_amdgcn_wmma_f32_16x16x32_f16(false, a, false, bm,
                                                       (short)0, acc[nt], false, false);
    }
  }
  if (RELU) {
    #pragma unroll
    for (int nt = 0; nt < NT; ++nt)
      #pragma unroll
      for (int i = 0; i < 8; ++i) acc[nt][i] = fmaxf(acc[nt][i], 0.0f);
  }
}

// C/D layout: acc[v] at lane L is element (M = v + 8*(L>=16), Ncol = L%16)
template<int NT>
__device__ __forceinline__ void store_acc_act(_Float16* act, int AS, int lane, const v8f* acc) {
  const int col0 = lane & 15, grp = lane >> 4;
  #pragma unroll
  for (int nt = 0; nt < NT; ++nt)
    #pragma unroll
    for (int v = 0; v < 8; ++v)
      act[(grp * 8 + v) * AS + nt * 16 + col0] = (_Float16)acc[nt][v];
}

// Weight staging: global -> LDS. Prefer CDNA5 async DMA path (no VGPR round-trip,
// tracked by ASYNCcnt, overlaps with the gather phase); fall back to load+ds_store.
__device__ __forceinline__ void stage_weights(_Float16* wlds, const _Float16* wg,
                                              int nh, int tid, int nthr) {
#if defined(USE_ASYNC_LDS)
  typedef __attribute__((address_space(1))) v4i* g4_t;
  typedef __attribute__((address_space(3))) v4i* l4_t;
  for (int i = tid * 8; i < nh; i += nthr * 8) {
    g4_t g = (g4_t)(wg + i);
    l4_t l = (l4_t)(wlds + i);
    __builtin_amdgcn_global_load_async_to_lds_b128(g, l, 0, 0);
  }
#else
  for (int i = tid * 8; i < nh; i += nthr * 8)
    *(h8*)(wlds + i) = *(const h8*)(wg + i);
#endif
}

__device__ __forceinline__ void wait_async0() {
#if defined(USE_ASYNC_LDS)
#  if defined(HAVE_WAIT_ASYNC_BUILTIN)
  __builtin_amdgcn_s_wait_asynccnt(0);
#  else
  asm volatile("s_wait_asynccnt 0x0" ::: "memory");
#  endif
#endif
}

// ---------- prep kernels ----------
// Convert W (Kin x Nout, f32 row-major) into B-fragment-ordered f16, K padded to Kpad.
// frag = kc*NT + nt ; halves h: K = kc*32 + (lane>=16)*16 + h, col = nt*16 + lane%16
__global__ void swizzle_w_kernel(const float* __restrict__ W, _Float16* __restrict__ out,
                                 int Kin, int Kpad, int Nout) {
  int NT = Nout >> 4;
  int total = (Kpad >> 5) * NT * 32;
  int t = blockIdx.x * blockDim.x + threadIdx.x;
  if (t >= total) return;
  int lane = t & 31, frag = t >> 5;
  int kc = frag / NT, nt = frag - kc * NT;
  int col = nt * 16 + (lane & 15);
  int kbase = kc * 32 + (lane >> 4) * 16;
  _Float16* o = out + (size_t)(frag * 32 + lane) * 16;
  #pragma unroll
  for (int h = 0; h < 16; ++h) {
    int K = kbase + h;
    float v = (K < Kin) ? W[(size_t)K * Nout + col] : 0.0f;
    o[h] = (_Float16)v;
  }
}

__global__ void cvt_f32_f16_kernel(const float* __restrict__ in, _Float16* __restrict__ out, int n) {
  int t = blockIdx.x * blockDim.x + threadIdx.x;
  if (t < n) out[t] = (_Float16)in[t];
}

__global__ void init_zero_kernel(unsigned* __restrict__ aggrMono, unsigned* __restrict__ deg, int N) {
  int t = blockIdx.x * blockDim.x + threadIdx.x;
  if (t < N * 64) aggrMono[t] = 0u;
  if (t < N) deg[t] = 0u;
}

__global__ void finalize_aggr_kernel(const unsigned* __restrict__ aggrMono,
                                     const unsigned* __restrict__ deg,
                                     _Float16* __restrict__ aggrH, int N) {
  int t = blockIdx.x * blockDim.x + threadIdx.x;
  if (t >= N * 64) return;
  int n = t >> 6;
  float v = (deg[n] > 0u) ? funmono(aggrMono[t]) : 0.0f;
  aggrH[t] = (_Float16)v;
}

// ---------- node input MLP: [64,128,128,64], f32 in -> f16 x0 out ----------
__global__ void __launch_bounds__(128) mlp_in_kernel(const float* __restrict__ x,
    const _Float16* __restrict__ W0, const float* __restrict__ b0,
    const _Float16* __restrict__ W1, const float* __restrict__ b1,
    const _Float16* __restrict__ W2, const float* __restrict__ b2,
    _Float16* __restrict__ x0h, int N) {
  extern __shared__ char smem[];
  _Float16* wlds = (_Float16*)smem;            // up to 16384 halves (32KB)
  _Float16* act  = (_Float16*)(smem + 32768);  // 4 waves * 16*128 halves (16KB)
  const int tid = threadIdx.x, lane = tid & 31, wv = tid >> 5;
  const int col0 = lane & 15, grp = lane >> 4;
  const int rowBase = blockIdx.x * 64 + wv * 16;
  _Float16* myact = act + wv * (16 * 128);
  stage_weights(wlds, W0, 64 * 128, tid, 128);   // async DMA overlaps with gather below
  { // stage 16x64 input rows (AS=128), f32->f16
    int n = rowBase + col0;
    _Float16* dp = myact + col0 * 128 + grp * 32;
    if (n < N) {
      const float* sp = x + (size_t)n * 64 + grp * 32;
      #pragma unroll
      for (int i = 0; i < 32; ++i) dp[i] = (_Float16)sp[i];
    } else {
      #pragma unroll
      for (int i = 0; i < 32; ++i) dp[i] = (_Float16)0.0f;
    }
  }
  v8f acc[8];
  wait_async0();
  __syncthreads();
  run_layer<2, 8, true>(myact, 128, wlds, b0, lane, acc);
  store_acc_act<8>(myact, 128, lane, acc);
  __syncthreads();
  stage_weights(wlds, W1, 128 * 128, tid, 128);
  wait_async0();
  __syncthreads();
  run_layer<4, 8, true>(myact, 128, wlds, b1, lane, acc);
  store_acc_act<8>(myact, 128, lane, acc);
  __syncthreads();
  stage_weights(wlds, W2, 128 * 64, tid, 128);
  wait_async0();
  __syncthreads();
  run_layer<4, 4, false>(myact, 128, wlds, b2, lane, acc);
  #pragma unroll
  for (int nt = 0; nt < 4; ++nt)
    #pragma unroll
    for (int v = 0; v < 8; ++v) {
      int n = rowBase + grp * 8 + v;
      if (n < N) x0h[(size_t)n * 64 + nt * 16 + col0] = (_Float16)acc[nt][v];
    }
}

// ---------- edge MLP: [160(pad of 148),128,128,64] + segment-max ----------
__global__ void __launch_bounds__(128) edge_kernel(const int* __restrict__ ei,
    const float* __restrict__ pos,
    const _Float16* __restrict__ x0h, const _Float16* __restrict__ semH,
    const _Float16* __restrict__ W0, const float* __restrict__ b0,
    const _Float16* __restrict__ W1, const float* __restrict__ b1,
    const _Float16* __restrict__ W2, const float* __restrict__ b2,
    unsigned* __restrict__ aggrMono, unsigned* __restrict__ deg, int E) {
  extern __shared__ char smem[];
  _Float16* wlds = (_Float16*)smem;                      // up to 20480 halves (40KB)
  _Float16* act  = (_Float16*)(smem + 40960);            // 4 waves * 16*160 halves (20KB)
  int*      dIdx = (int*)(smem + 40960 + 20480);         // 64 ints
  const int tid = threadIdx.x, lane = tid & 31, wv = tid >> 5;
  const int col0 = lane & 15, grp = lane >> 4;
  const int e0 = blockIdx.x * 64 + wv * 16;
  _Float16* myact = act + wv * (16 * 160);
  stage_weights(wlds, W0, 160 * 128, tid, 128);  // async DMA overlaps with gather below
  { // gather: cols [0..63]=x0[dst] [64..127]=x0[src] [128..135]=sem[dst]
    //         [136..143]=sem[src] [144..147]=rel [148..159]=0
    int r = col0, p = grp;
    int e = e0 + r;
    _Float16* ar = myact + r * 160;
    if (e < E) {
      int s = ei[e], d = ei[E + e];
      if (p == 0) {
        const h8* xs = (const h8*)(x0h + (size_t)d * 64);
        h8* dp = (h8*)ar;
        #pragma unroll
        for (int i = 0; i < 8; ++i) dp[i] = xs[i];
        *(h8*)(ar + 128) = *(const h8*)(semH + (size_t)d * 8);
        const float* pi = pos + (size_t)d * 4;
        const float* pj = pos + (size_t)s * 4;
        float hx = pi[2], hy = pi[3];
        float nrm = sqrtf(hx * hx + hy * hy);
        float c = hx / nrm, sn = hy / nrm;
        float dx = pj[0] - pi[0], dy = pj[1] - pi[1];
        ar[144] = (_Float16)nan0(c * dx + sn * dy);
        ar[145] = (_Float16)nan0(c * dy - sn * dx);
        ar[146] = (_Float16)nan0(c * pj[2] + sn * pj[3]);
        ar[147] = (_Float16)nan0(c * pj[3] - sn * pj[2]);
        ar[148] = ar[149] = ar[150] = ar[151] = (_Float16)0.0f;
        dIdx[wv * 16 + r] = d;
        atomicAdd(deg + d, 1u);
      } else {
        const h8* xs = (const h8*)(x0h + (size_t)s * 64);
        h8* dp = (h8*)(ar + 64);
        #pragma unroll
        for (int i = 0; i < 8; ++i) dp[i] = xs[i];
        *(h8*)(ar + 136) = *(const h8*)(semH + (size_t)s * 8);
        zero_h8(ar + 152);
      }
    } else {
      if (p == 0) {
        #pragma unroll
        for (int i = 0; i < 8; ++i) zero_h8(ar + i * 8);
        zero_h8(ar + 128); zero_h8(ar + 144);
        dIdx[wv * 16 + r] = -1;
      } else {
        #pragma unroll
        for (int i = 0; i < 8; ++i) zero_h8(ar + 64 + i * 8);
        zero_h8(ar + 136); zero_h8(ar + 152);
      }
    }
  }
  v8f acc[8];
  wait_async0();
  __syncthreads();
  run_layer<5, 8, true>(myact, 160, wlds, b0, lane, acc);
  store_acc_act<8>(myact, 160, lane, acc);
  __syncthreads();
  stage_weights(wlds, W1, 128 * 128, tid, 128);
  wait_async0();
  __syncthreads();
  run_layer<4, 8, true>(myact, 160, wlds, b1, lane, acc);
  store_acc_act<8>(myact, 160, lane, acc);
  __syncthreads();
  stage_weights(wlds, W2, 128 * 64, tid, 128);
  wait_async0();
  __syncthreads();
  run_layer<4, 4, false>(myact, 160, wlds, b2, lane, acc);
  // segment max into aggr via order-preserving uint atomicMax
  #pragma unroll
  for (int nt = 0; nt < 4; ++nt)
    #pragma unroll
    for (int v = 0; v < 8; ++v) {
      int m = grp * 8 + v;
      int e = e0 + m;
      if (e < E) {
        int d = dIdx[wv * 16 + m];
        atomicMax(aggrMono + (size_t)d * 64 + nt * 16 + col0, fmono(acc[nt][v]));
      }
    }
}

// ---------- update MLP: [160(pad of 136),128,64], f16 in -> f16 x1 ----------
__global__ void __launch_bounds__(128) mlp_upd_kernel(const _Float16* __restrict__ x0h,
    const _Float16* __restrict__ aggrH, const _Float16* __restrict__ semH,
    const _Float16* __restrict__ W0, const float* __restrict__ b0,
    const _Float16* __restrict__ W1, const float* __restrict__ b1,
    _Float16* __restrict__ x1h, int N) {
  extern __shared__ char smem[];
  _Float16* wlds = (_Float16*)smem;             // up to 20480 halves (40KB)
  _Float16* act  = (_Float16*)(smem + 40960);   // 4 waves * 16*160 halves (20KB)
  const int tid = threadIdx.x, lane = tid & 31, wv = tid >> 5;
  const int col0 = lane & 15, grp = lane >> 4;
  const int rowBase = blockIdx.x * 64 + wv * 16;
  _Float16* myact = act + wv * (16 * 160);
  stage_weights(wlds, W0, 160 * 128, tid, 128);
  { // cols: [0..63]=x0 [64..127]=aggr [128..135]=sem [136..159]=0
    int n = rowBase + col0;
    _Float16* ar = myact + col0 * 160;
    if (n < N) {
      if (grp == 0) {
        const h8* s = (const h8*)(x0h + (size_t)n * 64);
        h8* d = (h8*)ar;
        #pragma unroll
        for (int i = 0; i < 8; ++i) d[i] = s[i];
        *(h8*)(ar + 128) = *(const h8*)(semH + (size_t)n * 8);
      } else {
        const h8* s = (const h8*)(aggrH + (size_t)n * 64);
        h8* d = (h8*)(ar + 64);
        #pragma unroll
        for (int i = 0; i < 8; ++i) d[i] = s[i];
        zero_h8(ar + 136); zero_h8(ar + 144); zero_h8(ar + 152);
      }
    } else {
      if (grp == 0) {
        #pragma unroll
        for (int i = 0; i < 8; ++i) zero_h8(ar + i * 8);
        zero_h8(ar + 128);
      } else {
        #pragma unroll
        for (int i = 0; i < 8; ++i) zero_h8(ar + 64 + i * 8);
        zero_h8(ar + 136); zero_h8(ar + 144); zero_h8(ar + 152);
      }
    }
  }
  v8f acc[8];
  wait_async0();
  __syncthreads();
  run_layer<5, 8, true>(myact, 160, wlds, b0, lane, acc);
  store_acc_act<8>(myact, 160, lane, acc);
  __syncthreads();
  stage_weights(wlds, W1, 128 * 64, tid, 128);
  wait_async0();
  __syncthreads();
  run_layer<4, 4, false>(myact, 160, wlds, b1, lane, acc);
  #pragma unroll
  for (int nt = 0; nt < 4; ++nt)
    #pragma unroll
    for (int v = 0; v < 8; ++v) {
      int n = rowBase + grp * 8 + v;
      if (n < N) x1h[(size_t)n * 64 + nt * 16 + col0] = (_Float16)acc[nt][v];
    }
}

// ---------- output MLP: [64,128,128,64], f16 in -> f32 out ----------
__global__ void __launch_bounds__(128) mlp_out_kernel(const _Float16* __restrict__ x1h,
    const _Float16* __restrict__ W0, const float* __restrict__ b0,
    const _Float16* __restrict__ W1, const float* __restrict__ b1,
    const _Float16* __restrict__ W2, const float* __restrict__ b2,
    float* __restrict__ out, int N) {
  extern __shared__ char smem[];
  _Float16* wlds = (_Float16*)smem;
  _Float16* act  = (_Float16*)(smem + 32768);
  const int tid = threadIdx.x, lane = tid & 31, wv = tid >> 5;
  const int col0 = lane & 15, grp = lane >> 4;
  const int rowBase = blockIdx.x * 64 + wv * 16;
  _Float16* myact = act + wv * (16 * 128);
  stage_weights(wlds, W0, 64 * 128, tid, 128);
  {
    int n = rowBase + col0;
    _Float16* dp = myact + col0 * 128 + grp * 32;
    if (n < N) {
      const h8* sp = (const h8*)(x1h + (size_t)n * 64 + grp * 32);
      h8* d8 = (h8*)dp;
      #pragma unroll
      for (int i = 0; i < 4; ++i) d8[i] = sp[i];
    } else {
      #pragma unroll
      for (int i = 0; i < 4; ++i) zero_h8(dp + i * 8);
    }
  }
  v8f acc[8];
  wait_async0();
  __syncthreads();
  run_layer<2, 8, true>(myact, 128, wlds, b0, lane, acc);
  store_acc_act<8>(myact, 128, lane, acc);
  __syncthreads();
  stage_weights(wlds, W1, 128 * 128, tid, 128);
  wait_async0();
  __syncthreads();
  run_layer<4, 8, true>(myact, 128, wlds, b1, lane, acc);
  store_acc_act<8>(myact, 128, lane, acc);
  __syncthreads();
  stage_weights(wlds, W2, 128 * 64, tid, 128);
  wait_async0();
  __syncthreads();
  run_layer<4, 4, false>(myact, 128, wlds, b2, lane, acc);
  #pragma unroll
  for (int nt = 0; nt < 4; ++nt)
    #pragma unroll
    for (int v = 0; v < 8; ++v) {
      int n = rowBase + grp * 8 + v;
      if (n < N) out[(size_t)n * 64 + nt * 16 + col0] = acc[nt][v];
    }
}

// ---------- launcher ----------
extern "C" void kernel_launch(void* const* d_in, const int* in_sizes, int n_in,
                              void* d_out, int out_size, void* d_ws, size_t ws_size,
                              hipStream_t stream) {
  const int N = in_sizes[0] / 64;
  const int E = in_sizes[1] / 2;

  const float* x   = (const float*)d_in[0];
  const int*   ei  = (const int*)d_in[1];
  const float* pos = (const float*)d_in[2];
  const float* sem = (const float*)d_in[3];
  const float* inW0 = (const float*)d_in[4];  const float* inB0 = (const float*)d_in[5];
  const float* inW1 = (const float*)d_in[6];  const float* inB1 = (const float*)d_in[7];
  const float* inW2 = (const float*)d_in[8];  const float* inB2 = (const float*)d_in[9];
  const float* egW0 = (const float*)d_in[10]; const float* egB0 = (const float*)d_in[11];
  const float* egW1 = (const float*)d_in[12]; const float* egB1 = (const float*)d_in[13];
  const float* egW2 = (const float*)d_in[14]; const float* egB2 = (const float*)d_in[15];
  const float* upW0 = (const float*)d_in[16]; const float* upB0 = (const float*)d_in[17];
  const float* upW1 = (const float*)d_in[18]; const float* upB1 = (const float*)d_in[19];
  const float* otW0 = (const float*)d_in[20]; const float* otB0 = (const float*)d_in[21];
  const float* otW1 = (const float*)d_in[22]; const float* otB1 = (const float*)d_in[23];
  const float* otW2 = (const float*)d_in[24]; const float* otB2 = (const float*)d_in[25];

  char* ws = (char*)d_ws;
  size_t off = 0;
  auto alloc = [&](size_t bytes) -> char* {
    char* p = ws + off;
    off = (off + bytes + 255) & ~(size_t)255;
    return p;
  };
  _Float16* inW0h = (_Float16*)alloc((size_t)64 * 128 * 2);
  _Float16* inW1h = (_Float16*)alloc((size_t)128 * 128 * 2);
  _Float16* inW2h = (_Float16*)alloc((size_t)128 * 64 * 2);
  _Float16* egW0h = (_Float16*)alloc((size_t)160 * 128 * 2);
  _Float16* egW1h = (_Float16*)alloc((size_t)128 * 128 * 2);
  _Float16* egW2h = (_Float16*)alloc((size_t)128 * 64 * 2);
  _Float16* upW0h = (_Float16*)alloc((size_t)160 * 128 * 2);
  _Float16* upW1h = (_Float16*)alloc((size_t)128 * 64 * 2);
  _Float16* otW0h = (_Float16*)alloc((size_t)64 * 128 * 2);
  _Float16* otW1h = (_Float16*)alloc((size_t)128 * 128 * 2);
  _Float16* otW2h = (_Float16*)alloc((size_t)128 * 64 * 2);
  _Float16* x0h   = (_Float16*)alloc((size_t)N * 64 * 2);
  _Float16* semH  = (_Float16*)alloc((size_t)N * 8 * 2);
  unsigned* aggrM = (unsigned*)alloc((size_t)N * 64 * 4);
  unsigned* deg   = (unsigned*)alloc((size_t)N * 4);
  _Float16* aggrH = (_Float16*)alloc((size_t)N * 64 * 2);
  _Float16* x1h   = (_Float16*)alloc((size_t)N * 64 * 2);
  (void)ws_size; (void)n_in; (void)out_size;

  auto swz = [&](const float* W, _Float16* o, int Kin, int Kpad, int Nout) {
    int total = (Kpad / 32) * (Nout / 16) * 32;
    swizzle_w_kernel<<<(total + 255) / 256, 256, 0, stream>>>(W, o, Kin, Kpad, Nout);
  };
  swz(inW0, inW0h, 64, 64, 128);
  swz(inW1, inW1h, 128, 128, 128);
  swz(inW2, inW2h, 128, 128, 64);
  swz(egW0, egW0h, 148, 160, 128);
  swz(egW1, egW1h, 128, 128, 128);
  swz(egW2, egW2h, 128, 128, 64);
  swz(upW0, upW0h, 136, 160, 128);
  swz(upW1, upW1h, 128, 128, 64);
  swz(otW0, otW0h, 64, 64, 128);
  swz(otW1, otW1h, 128, 128, 128);
  swz(otW2, otW2h, 128, 128, 64);

  cvt_f32_f16_kernel<<<(N * 8 + 255) / 256, 256, 0, stream>>>(sem, semH, N * 8);
  init_zero_kernel<<<(N * 64 + 255) / 256, 256, 0, stream>>>(aggrM, deg, N);

  const int nodeBlocks = (N + 63) / 64;
  const int edgeBlocks = (E + 63) / 64;
  const size_t SM_NODE = 32768 + 4 * 16 * 128 * 2;            // 49152
  const size_t SM_UPD  = 40960 + 4 * 16 * 160 * 2;            // 61440
  const size_t SM_EDGE = 40960 + 4 * 16 * 160 * 2 + 64 * 4;   // 61696

  mlp_in_kernel<<<nodeBlocks, 128, SM_NODE, stream>>>(
      x, inW0h, inB0, inW1h, inB1, inW2h, inB2, x0h, N);

  edge_kernel<<<edgeBlocks, 128, SM_EDGE, stream>>>(
      ei, pos, x0h, semH, egW0h, egB0, egW1h, egB1, egW2h, egB2, aggrM, deg, E);

  finalize_aggr_kernel<<<(N * 64 + 255) / 256, 256, 0, stream>>>(aggrM, deg, aggrH, N);

  mlp_upd_kernel<<<nodeBlocks, 128, SM_UPD, stream>>>(
      x0h, aggrH, semH, upW0h, upB0, upW1h, upB1, x1h, N);

  mlp_out_kernel<<<nodeBlocks, 128, SM_NODE, stream>>>(
      x1h, otW0h, otB0, otW1h, otB1, otW2h, otB2, (float*)d_out, N);
}
